// GNN_27307402068304
// MI455X (gfx1250) — compile-verified, matrix operand order checked by reading, not run on hardware
//
#include <hip/hip_runtime.h>
#include <hip/hip_bf16.h>

typedef __attribute__((ext_vector_type(16))) _Float16 v16h;
typedef __attribute__((ext_vector_type(8)))  float    v8f;

#define EMB 128

// ---------- helpers ----------
__device__ __forceinline__ unsigned encF(float f) {
    unsigned u = __float_as_uint(f);
    return (u & 0x80000000u) ? ~u : (u | 0x80000000u);
}
__device__ __forceinline__ float decF(unsigned u) {
    unsigned b = (u & 0x80000000u) ? (u & 0x7FFFFFFFu) : ~u;
    return __uint_as_float(b);
}
__device__ __forceinline__ void edge_sd(const int* ei, int e, int E, int& s, int& d) {
    if (e < E) { s = ei[e]; d = ei[E + e]; }
    else       { int i = e - E; s = i; d = i; }   // self loops appended
}

// ---------- tiny reductions / setup ----------
__global__ void k_dots(const float* W0, const float* as0, const float* ad0, float* S) {
    __shared__ float sh0[EMB], sh1[EMB];
    int c = threadIdx.x;
    sh0[c] = W0[c] * as0[c];
    sh1[c] = W0[c] * ad0[c];
    __syncthreads();
    for (int o = 64; o > 0; o >>= 1) {
        if (c < o) { sh0[c] += sh0[c + o]; sh1[c] += sh1[c + o]; }
        __syncthreads();
    }
    if (c == 0) { S[0] = sh0[0]; S[1] = sh1[0]; }
}

__global__ void k_node_scale(const float* x, const float* S, float* asn, float* adn, int N) {
    int i = blockIdx.x * blockDim.x + threadIdx.x;
    if (i >= N) return;
    float v = x[i];
    asn[i] = v * S[0];
    adn[i] = v * S[1];
}

// ---------- edge softmax (shared by both layers) ----------
__global__ void k_edge_logit(const float* asn, const float* adn, const int* ei,
                             float* logits, unsigned* mEnc, int E, int Etot) {
    int e = blockIdx.x * blockDim.x + threadIdx.x;
    if (e >= Etot) return;
    int s, d; edge_sd(ei, e, E, s, d);
    float v = asn[s] + adn[d];
    v = (v > 0.f) ? v : 0.2f * v;          // leaky_relu(0.2)
    logits[e] = v;
    atomicMax(&mEnc[d], encF(v));
}

__global__ void k_edge_exp(const int* ei, float* logits, const unsigned* mEnc,
                           float* denom, int E, int Etot) {
    int e = blockIdx.x * blockDim.x + threadIdx.x;
    if (e >= Etot) return;
    int s, d; edge_sd(ei, e, E, s, d);
    float ex = __expf(logits[e] - decF(mEnc[d]));
    logits[e] = ex;                         // overwrite with exp value
    atomicAdd(&denom[d], ex);
}

__global__ void k_edge_agg_scalar(const float* x, const int* ei, const float* ex,
                                  const float* denom, float* sacc, int E, int Etot) {
    int e = blockIdx.x * blockDim.x + threadIdx.x;
    if (e >= Etot) return;
    int s, d; edge_sd(ei, e, E, s, d);
    float alpha = ex[e] / (denom[d] + 1e-16f);
    atomicAdd(&sacc[d], alpha * x[s]);
}

__global__ void k_edge_agg_vec(const float* hh, const int* ei, const float* ex,
                               const float* denom, float* out, int E, int Etot) {
    long long t = (long long)blockIdx.x * blockDim.x + threadIdx.x;
    int e = (int)(t >> 5);
    if (e >= Etot) return;
    int grp = (int)(t & 31);                // 4 channels per lane, 32 lanes per edge
    int s, d; edge_sd(ei, e, E, s, d);
    float alpha = ex[e] / (denom[d] + 1e-16f);
    float4 v = ((const float4*)(hh + (size_t)s * EMB))[grp];
    float* o = out + (size_t)d * EMB + grp * 4;
    atomicAdd(o + 0, alpha * v.x);
    atomicAdd(o + 1, alpha * v.y);
    atomicAdd(o + 2, alpha * v.z);
    atomicAdd(o + 3, alpha * v.w);
}

// ---------- layer-1 post, norms ----------
__global__ void k_l1_post(const float* sacc, const float* W0, const float* b0, float* h, int N) {
    int t = blockIdx.x * blockDim.x + threadIdx.x;
    if (t >= N * EMB) return;
    int i = t >> 7, c = t & 127;
    h[t] = tanhf(sacc[i] * W0[c] + b0[c]);
}

__global__ void k_bn_stats(const float* h, float* bnSum, float* bnSq, int N) {
    int c = threadIdx.x;                    // 128 threads = channels
    float s = 0.f, q = 0.f;
    for (int i = blockIdx.x; i < N; i += gridDim.x) {
        float v = h[(size_t)i * EMB + c];
        s += v; q += v * v;
    }
    atomicAdd(&bnSum[c], s);
    atomicAdd(&bnSq[c], q);
}

__global__ void k_cnt(const int* batch, float* cnt, int N) {
    int i = blockIdx.x * blockDim.x + threadIdx.x;
    if (i >= N) return;
    atomicAdd(&cnt[batch[i]], 1.0f);
}

__global__ void k_bn_apply(float* h, const int* batch, const float* bnSum, const float* bnSq,
                           const float* bn_g, const float* bn_b, float* gsum, int N) {
    int t = blockIdx.x * blockDim.x + threadIdx.x;
    if (t >= N * EMB) return;
    int i = t >> 7, c = t & 127;
    float mu  = bnSum[c] / (float)N;
    float var = bnSq[c] / (float)N - mu * mu;
    float v = bn_g[c] * (h[t] - mu) * rsqrtf(var + 1e-5f) + bn_b[c];
    h[t] = v;
    atomicAdd(&gsum[(size_t)batch[i] * EMB + c], v);
}

__global__ void k_gn_center(float* h, const int* batch, const float* gsum, const float* cnt,
                            const float* gn_ms, float* gsq, int N) {
    int t = blockIdx.x * blockDim.x + threadIdx.x;
    if (t >= N * EMB) return;
    int i = t >> 7, c = t & 127;
    int g = batch[i];
    float gm = gsum[(size_t)g * EMB + c] / cnt[g];
    float v = h[t] - gm * gn_ms[c];
    h[t] = v;
    atomicAdd(&gsq[(size_t)g * EMB + c], v * v);
}

__global__ void k_gn_apply(const float* h, const int* batch, const float* gsq, const float* cnt,
                           const float* gn_w, const float* gn_b, _Float16* hH, int N) {
    int t = blockIdx.x * blockDim.x + threadIdx.x;
    if (t >= N * EMB) return;
    int i = t >> 7, c = t & 127;
    int g = batch[i];
    float gv = gsq[(size_t)g * EMB + c] / cnt[g];
    float v = gn_w[c] * h[t] * rsqrtf(gv + 1e-5f) + gn_b[c];
    hH[t] = (_Float16)v;
}

// ---------- weight transposes to f16 (n-major so B loads are contiguous) ----------
__global__ void k_w1t(const float* W1, _Float16* W1T) {
    int t = blockIdx.x * blockDim.x + threadIdx.x;
    if (t >= EMB * EMB) return;
    int k = t >> 7, n = t & 127;
    W1T[(size_t)n * EMB + k] = (_Float16)W1[t];
}

__global__ void k_woutt(const float* Wout, _Float16* WoutT, int K2, int OUT) {
    int t = blockIdx.x * blockDim.x + threadIdx.x;
    if (t >= K2 * OUT) return;
    int k = t / OUT, n = t % OUT;
    WoutT[(size_t)n * K2 + k] = (_Float16)Wout[t];
}

// ---------- WMMA GEMM 1: hh[N,128] = hH[N,128] @ W1 (via W1T n-major) ----------
__global__ void __launch_bounds__(256)
k_gemm1(const _Float16* A, const _Float16* BT, float* C, int N) {
    int wave = threadIdx.x >> 5;            // 8 waves -> 8 N-tiles (EMB/16)
    int lane = threadIdx.x & 31;
    int l16 = lane & 15, kh = lane >> 4;
    int mrow = blockIdx.x * 16 + l16;
    int ncol = wave * 16 + l16;
    int mr = (mrow < N) ? mrow : (N - 1);
    const _Float16* ap = A  + (size_t)mr   * EMB;
    const _Float16* bp = BT + (size_t)ncol * EMB;
    v8f acc = {};
    #pragma unroll
    for (int kk = 0; kk < EMB; kk += 32) {
        v16h a, b;
        #pragma unroll
        for (int j = 0; j < 8; ++j) {
            a[j]     = ap[kk +      kh * 8 + j];
            a[j + 8] = ap[kk + 16 + kh * 8 + j];
            b[j]     = bp[kk +      kh * 8 + j];
            b[j + 8] = bp[kk + 16 + kh * 8 + j];
        }
        acc = __builtin_amdgcn_wmma_f32_16x16x32_f16(false, a, false, b,
                                                     (short)0, acc, false, false);
    }
    // D layout: lane row-base = 8*kh, rows r..r+7 stride EMB -> one base, 8 stores
    float* cp = C + ((size_t)(blockIdx.x * 16 + 8 * kh) * EMB + wave * 16 + l16);
    if (blockIdx.x * 16 + 16 <= N) {        // full tile: unguarded incremental stores
        #pragma unroll
        for (int r = 0; r < 8; ++r) cp[(size_t)r * EMB] = acc[r];
    } else {                                // boundary tile: per-lane predication
        #pragma unroll
        for (int r = 0; r < 8; ++r) {
            int row = blockIdx.x * 16 + 8 * kh + r;
            if (row < N) cp[(size_t)r * EMB] = acc[r];
        }
    }
}

// ---------- layer-2 attention scalars ----------
__global__ void k_attdots(const float* hh, const float* as1, const float* ad1,
                          float* aS, float* aD, int N) {
    int i = blockIdx.x * blockDim.x + threadIdx.x;
    if (i >= N) return;
    const float4* p  = (const float4*)(hh + (size_t)i * EMB);
    const float4* pa = (const float4*)as1;
    const float4* pd = (const float4*)ad1;
    float s = 0.f, d = 0.f;
    #pragma unroll 8
    for (int j = 0; j < EMB / 4; ++j) {
        float4 v = p[j], a = pa[j], b = pd[j];
        s += v.x * a.x + v.y * a.y + v.z * a.z + v.w * a.w;
        d += v.x * b.x + v.y * b.y + v.z * b.z + v.w * b.w;
    }
    aS[i] = s; aD[i] = d;
}

// ---------- layer-2 post + pooling ----------
__global__ void k_l2_post(float* out2, const float* b1, const int* batch,
                          unsigned* pmaxE, float* psum, int N) {
    int t = blockIdx.x * blockDim.x + threadIdx.x;
    if (t >= N * EMB) return;
    int i = t >> 7, c = t & 127;
    float v = tanhf(out2[t] + b1[c]);
    out2[t] = v;
    int g = batch[i];
    atomicMax(&pmaxE[(size_t)g * EMB + c], encF(v));
    atomicAdd(&psum[(size_t)g * EMB + c], v);
}

__global__ void k_pool(const unsigned* pmaxE, const float* psum, const float* cnt,
                       _Float16* pooledH, int G) {
    int t = blockIdx.x * blockDim.x + threadIdx.x;
    if (t >= G * EMB) return;
    int g = t >> 7, c = t & 127;
    pooledH[(size_t)g * (2 * EMB) + c]       = (_Float16)decF(pmaxE[t]);
    pooledH[(size_t)g * (2 * EMB) + EMB + c] = (_Float16)(psum[t] / cnt[g]);
}

// ---------- WMMA GEMM 2: out[G,OUT] = pooled[G,256] @ Wout + bout ----------
__global__ void __launch_bounds__(256)
k_gemm2(const _Float16* A, const _Float16* BT, const float* bout,
        float* out, int G, int OUT) {
    int wave = threadIdx.x >> 5;
    int lane = threadIdx.x & 31;
    int l16 = lane & 15, kh = lane >> 4;
    int ntiles = (OUT + 15) / 16;
    int ntile = blockIdx.y * 8 + wave;
    bool wvalid = (ntile < ntiles);
    if (!wvalid) ntile = 0;                 // clamp: keep EXEC all-ones across WMMA
    int mrow = blockIdx.x * 16 + l16;
    int ncol = ntile * 16 + l16;
    int mr = (mrow < G) ? mrow : (G - 1);
    bool bvalid = (ncol < OUT);
    const _Float16* ap = A  + (size_t)mr * 256;
    const _Float16* bp = BT + (size_t)(bvalid ? ncol : 0) * 256;
    v8f acc = {};
    #pragma unroll
    for (int kk = 0; kk < 256; kk += 32) {
        v16h a, b;
        #pragma unroll
        for (int j = 0; j < 8; ++j) {
            a[j]     = ap[kk +      kh * 8 + j];
            a[j + 8] = ap[kk + 16 + kh * 8 + j];
            _Float16 b0 = bvalid ? bp[kk +      kh * 8 + j] : (_Float16)0.f;
            _Float16 b1 = bvalid ? bp[kk + 16 + kh * 8 + j] : (_Float16)0.f;
            b[j] = b0; b[j + 8] = b1;
        }
        acc = __builtin_amdgcn_wmma_f32_16x16x32_f16(false, a, false, b,
                                                     (short)0, acc, false, false);
    }
    int rowBase = blockIdx.x * 16 + 8 * kh;
    int col = ntile * 16 + l16;
    bool mfull = (blockIdx.x * 16 + 16 <= G);
    // single divergent predicate around all 8 stores (one EXEC toggle)
    if (wvalid && col < OUT && mfull) {
        float bc = bout[col];
        float* op = out + (size_t)rowBase * OUT + col;
        #pragma unroll
        for (int r = 0; r < 8; ++r) op[(size_t)r * OUT] = acc[r] + bc;
    } else if (wvalid && col < OUT) {
        float bc = bout[col];
        float* op = out + (size_t)rowBase * OUT + col;
        #pragma unroll
        for (int r = 0; r < 8; ++r)
            if (rowBase + r < G) op[(size_t)r * OUT] = acc[r] + bc;
    }
}

// ====================== host launch ======================
static inline void* bump(char*& p, size_t bytes) {
    void* r = (void*)p;
    p += (bytes + 255) & ~(size_t)255;
    return r;
}

extern "C" void kernel_launch(void* const* d_in, const int* in_sizes, int n_in,
                              void* d_out, int out_size, void* d_ws, size_t ws_size,
                              hipStream_t stream) {
    const float* x    = (const float*)d_in[0];
    const int*   ei   = (const int*)d_in[1];
    const int*   batch= (const int*)d_in[2];
    const float* W0   = (const float*)d_in[4];
    const float* as0  = (const float*)d_in[5];
    const float* ad0  = (const float*)d_in[6];
    const float* b0   = (const float*)d_in[7];
    const float* W1   = (const float*)d_in[8];
    const float* as1  = (const float*)d_in[9];
    const float* ad1  = (const float*)d_in[10];
    const float* b1   = (const float*)d_in[11];
    const float* bn_g = (const float*)d_in[12];
    const float* bn_b = (const float*)d_in[13];
    const float* gn_w = (const float*)d_in[14];
    const float* gn_b = (const float*)d_in[15];
    const float* gn_ms= (const float*)d_in[16];
    const float* Wout = (const float*)d_in[17];
    const float* bout = (const float*)d_in[18];

    const int N    = in_sizes[0];
    const int E    = in_sizes[1] / 2;
    const int Etot = E + N;
    const int OUT  = in_sizes[18];
    const int G    = out_size / OUT;

    // ---- workspace carve ----
    char* p = (char*)d_ws;
    float*    S      = (float*)   bump(p, 2 * sizeof(float));
    unsigned* mEnc   = (unsigned*)bump(p, (size_t)N * 4);
    float*    denom  = (float*)   bump(p, (size_t)N * 4);
    float*    logits = (float*)   bump(p, (size_t)Etot * 4);
    float*    asn    = (float*)   bump(p, (size_t)N * 4);
    float*    adn    = (float*)   bump(p, (size_t)N * 4);
    float*    sacc   = (float*)   bump(p, (size_t)N * 4);
    float*    h      = (float*)   bump(p, (size_t)N * EMB * 4);
    float*    bnSum  = (float*)   bump(p, EMB * 4);
    float*    bnSq   = (float*)   bump(p, EMB * 4);
    float*    cnt    = (float*)   bump(p, (size_t)G * 4);
    float*    gsum   = (float*)   bump(p, (size_t)G * EMB * 4);
    float*    gsq    = (float*)   bump(p, (size_t)G * EMB * 4);
    _Float16* hH     = (_Float16*)bump(p, (size_t)N * EMB * 2);
    _Float16* W1T    = (_Float16*)bump(p, (size_t)EMB * EMB * 2);
    float*    hh     = (float*)   bump(p, (size_t)N * EMB * 4);
    float*    aS1    = (float*)   bump(p, (size_t)N * 4);
    float*    aD1    = (float*)   bump(p, (size_t)N * 4);
    float*    out2   = (float*)   bump(p, (size_t)N * EMB * 4);
    unsigned* pmaxE  = (unsigned*)bump(p, (size_t)G * EMB * 4);
    float*    psum   = (float*)   bump(p, (size_t)G * EMB * 4);
    _Float16* pooledH= (_Float16*)bump(p, (size_t)G * 2 * EMB * 2);
    _Float16* WoutT  = (_Float16*)bump(p, (size_t)OUT * 256 * 2);

    const int TB = 256;
    dim3 blk(TB);
    auto cdiv = [](long long a, long long b) { return (int)((a + b - 1) / b); };

    // ---- zero accumulators (encF init 0 is below every finite encoding) ----
    hipMemsetAsync(mEnc,  0, (size_t)N * 4, stream);
    hipMemsetAsync(denom, 0, (size_t)N * 4, stream);
    hipMemsetAsync(sacc,  0, (size_t)N * 4, stream);
    hipMemsetAsync(bnSum, 0, EMB * 4, stream);
    hipMemsetAsync(bnSq,  0, EMB * 4, stream);
    hipMemsetAsync(cnt,   0, (size_t)G * 4, stream);
    hipMemsetAsync(gsum,  0, (size_t)G * EMB * 4, stream);
    hipMemsetAsync(gsq,   0, (size_t)G * EMB * 4, stream);

    // ---- layer 1 (rank-1 GAT -> scalar aggregation) ----
    k_dots<<<1, EMB, 0, stream>>>(W0, as0, ad0, S);
    k_node_scale<<<cdiv(N, TB), blk, 0, stream>>>(x, S, asn, adn, N);
    k_edge_logit<<<cdiv(Etot, TB), blk, 0, stream>>>(asn, adn, ei, logits, mEnc, E, Etot);
    k_edge_exp<<<cdiv(Etot, TB), blk, 0, stream>>>(ei, logits, mEnc, denom, E, Etot);
    k_edge_agg_scalar<<<cdiv(Etot, TB), blk, 0, stream>>>(x, ei, logits, denom, sacc, E, Etot);
    k_l1_post<<<cdiv((long long)N * EMB, TB), blk, 0, stream>>>(sacc, W0, b0, h, N);

    // ---- BatchNorm + GraphNorm ----
    k_bn_stats<<<512, EMB, 0, stream>>>(h, bnSum, bnSq, N);
    k_cnt<<<cdiv(N, TB), blk, 0, stream>>>(batch, cnt, N);
    k_bn_apply<<<cdiv((long long)N * EMB, TB), blk, 0, stream>>>(h, batch, bnSum, bnSq, bn_g, bn_b, gsum, N);
    k_gn_center<<<cdiv((long long)N * EMB, TB), blk, 0, stream>>>(h, batch, gsum, cnt, gn_ms, gsq, N);
    k_gn_apply<<<cdiv((long long)N * EMB, TB), blk, 0, stream>>>(h, batch, gsq, cnt, gn_w, gn_b, hH, N);

    // ---- layer 2 feature transform via WMMA ----
    k_w1t<<<cdiv(EMB * EMB, TB), blk, 0, stream>>>(W1, W1T);
    k_gemm1<<<cdiv(N, 16), blk, 0, stream>>>(hH, W1T, hh, N);
    k_attdots<<<cdiv(N, TB), blk, 0, stream>>>(hh, as1, ad1, aS1, aD1, N);

    // ---- layer 2 edge softmax + full-width aggregation ----
    hipMemsetAsync(mEnc,  0, (size_t)N * 4, stream);
    hipMemsetAsync(denom, 0, (size_t)N * 4, stream);
    hipMemsetAsync(out2,  0, (size_t)N * EMB * 4, stream);
    hipMemsetAsync(pmaxE, 0, (size_t)G * EMB * 4, stream);
    hipMemsetAsync(psum,  0, (size_t)G * EMB * 4, stream);
    k_edge_logit<<<cdiv(Etot, TB), blk, 0, stream>>>(aS1, aD1, ei, logits, mEnc, E, Etot);
    k_edge_exp<<<cdiv(Etot, TB), blk, 0, stream>>>(ei, logits, mEnc, denom, E, Etot);
    k_edge_agg_vec<<<cdiv((long long)Etot * 32, TB), blk, 0, stream>>>(hh, ei, logits, denom, out2, E, Etot);

    // ---- tanh + pooling ----
    k_l2_post<<<cdiv((long long)N * EMB, TB), blk, 0, stream>>>(out2, b1, batch, pmaxE, psum, N);
    k_pool<<<cdiv((long long)G * EMB, TB), blk, 0, stream>>>(pmaxE, psum, cnt, pooledH, G);

    // ---- head GEMM via WMMA ----
    k_woutt<<<cdiv((long long)256 * OUT, TB), blk, 0, stream>>>(Wout, WoutT, 256, OUT);
    int ntiles = (OUT + 15) / 16;
    dim3 grid2(cdiv(G, 16), cdiv(ntiles, 8));
    k_gemm2<<<grid2, blk, 0, stream>>>(pooledH, WoutT, bout, (float*)d_out, G, OUT);
}
